// SkeletalDiscBlock_79714593014262
// MI455X (gfx1250) — matrix operand order repaired and checked by reading, not compiled, not run on hardware
//
#include <hip/hip_runtime.h>

// ---------------- problem constants ----------------
constexpr int kJ   = 24;
constexpr int kD   = 4;
constexpr int kCIN = 32;
constexpr int kCOUT= 32;
constexpr int kKW  = 3;
constexpr int kB   = 8;
constexpr int kT   = 8192;
constexpr int kP   = 12;
constexpr float kEPS   = 1e-5f;
constexpr float kSLOPE = 0.2f;

constexpr int kCHY  = kJ * kCOUT;   // 768 conv output channels
constexpr int kKDIM = kD * kCIN;    // 128 reduction channels per joint
constexpr int kTN   = 64;           // time tile per block
constexpr int kXROWS = kTN + 2;     // halo rows (conv pad=1)
constexpr int kXSTRIDE = kKDIM + 8; // 136 (x2B = 272B, multiple of 16B)

// ---------------- vector types ----------------
typedef __attribute__((ext_vector_type(8)))  unsigned short u16x8;
typedef __attribute__((ext_vector_type(16))) unsigned short u16x16;
typedef __attribute__((ext_vector_type(16))) __bf16         v16bf;
typedef __attribute__((ext_vector_type(8)))  float          v8f;
typedef __attribute__((ext_vector_type(4)))  float          f32x4;

__device__ __forceinline__ unsigned short f2bf(float f) {
  unsigned u = __float_as_uint(f);
  u += 0x7FFFu + ((u >> 16) & 1u);          // round-to-nearest-even
  return (unsigned short)(u >> 16);
}
__device__ __forceinline__ float bf2f(unsigned short h) {
  return __uint_as_float(((unsigned)h) << 16);
}

// ============================================================
// Kernel 1: grouped conv as BF16 WMMA GEMM.
// Block = (batch b, joint j, 64-wide t tile). 8 waves -> 32x64 output tile.
// y[b, j*32+m, t] = sum_{sh,kd} W[j*32+m, kd, sh] * xg[b, j, kd, t+sh-1]
// Bias is intentionally skipped: BatchNorm's mean subtraction cancels it.
// ============================================================
__global__ __launch_bounds__(256)
void conv_wmma_kernel(const float* __restrict__ x, const float* __restrict__ W,
                      const int* __restrict__ adj, unsigned short* __restrict__ ybf) {
  __shared__ __align__(16) unsigned short xS[kXROWS * kXSTRIDE];     // [t_local][d*32+c]
  __shared__ __align__(16) unsigned short wS[kKW * kCOUT * kKDIM];   // [shift][m][kd]

  const int bid  = blockIdx.x;
  const int tile = bid & (kT / kTN - 1);          // 128 tiles
  const int j    = (bid >> 7) % kJ;
  const int b    = bid / ((kT / kTN) * kJ);
  const int t0   = tile * kTN;
  const int tid  = threadIdx.x;

  // ---- stage x (gathered via adj, transposed, zero-padded halo) ----
  for (int e = tid; e < kKDIM * kXROWS; e += 256) {
    int ch = e / kXROWS;                 // d*32+c
    int tt = e - ch * kXROWS;            // 0..65  (coalesced along t)
    int d  = ch >> 5;
    int c  = ch & 31;
    int ja = adj[j * 4 + d];
    int tg = t0 - 1 + tt;
    float v = 0.f;
    if (tg >= 0 && tg < kT)
      v = x[(((size_t)b * kJ + ja) * kCIN + c) * kT + tg];
    xS[tt * kXSTRIDE + ch] = f2bf(v);
  }
  // ---- stage W_j as bf16: wS[sh][m][kd] ----
  for (int e = tid; e < kKW * kCOUT * kKDIM; e += 256) {
    int sh = e >> 12;                    // /4096
    int m  = (e >> 7) & 31;
    int kd = e & 127;
    wS[e] = f2bf(W[((size_t)(j * kCOUT + m) * kKDIM + kd) * kKW + sh]);
  }
  __syncthreads();

  const int wave = tid >> 5, lane = tid & 31;
  const int rh = wave & 1;               // which 16-row half of the 32 channels
  const int nt = wave >> 1;              // which 16-col slice of the 64 t's
  const int lg = lane >> 4;              // lane group 0/1
  const int ln = lane & 15;

  v8f acc = {0.f, 0.f, 0.f, 0.f, 0.f, 0.f, 0.f, 0.f};
  #pragma unroll
  for (int sh = 0; sh < kKW; ++sh) {
    const int trow = nt * 16 + ln + sh;  // local t of this lane's B column (+halo shift)
    #pragma unroll
    for (int kb = 0; kb < kKDIM / 32; ++kb) {
      // A fragment (16x32 bf16): lanes 0-15 hold K{0..7,16..23}, lanes 16-31 K{8..15,24..31}
      const unsigned short* wrow = &wS[(sh * kCOUT + rh * 16 + ln) * kKDIM + kb * 32];
      u16x8 alo = *(const u16x8*)(wrow + 8 * lg);
      u16x8 ahi = *(const u16x8*)(wrow + 16 + 8 * lg);
      // B fragment (32x16 bf16): lanes 0-15 hold K{0..7,16..23}, lanes 16-31 K{8..15,24..31}; col = ln
      const unsigned short* xrow = &xS[trow * kXSTRIDE + kb * 32];
      u16x8 blo = *(const u16x8*)(xrow + 8 * lg);
      u16x8 bhi = *(const u16x8*)(xrow + 16 + 8 * lg);
      v16bf av = __builtin_bit_cast(v16bf,
          __builtin_shufflevector(alo, ahi, 0,1,2,3,4,5,6,7,8,9,10,11,12,13,14,15));
      v16bf bv = __builtin_bit_cast(v16bf,
          __builtin_shufflevector(blo, bhi, 0,1,2,3,4,5,6,7,8,9,10,11,12,13,14,15));
      acc = __builtin_amdgcn_wmma_f32_16x16x32_bf16(false, av, false, bv,
                                                    (short)0, acc, false, false);
    }
  }

  // C layout: VGPR r, lane -> (m = r + 8*lg, n = ln)
  const int tg = t0 + nt * 16 + ln;
  const size_t base = ((size_t)b * kCHY + j * kCOUT) * kT;
  #pragma unroll
  for (int r = 0; r < 8; ++r) {
    int mo = rh * 16 + r + 8 * lg;
    ybf[base + (size_t)mo * kT + tg] = f2bf(acc[r]);
  }
}

// ============================================================
// Kernel 2: per-channel BN statistics over (B,T) -> scale/shift.
// ============================================================
__global__ __launch_bounds__(256)
void bn_stats_kernel(const unsigned short* __restrict__ ybf,
                     const float* __restrict__ gamma, const float* __restrict__ beta,
                     float* __restrict__ scale, float* __restrict__ shift) {
  __shared__ float sred[256];
  __shared__ float qred[256];
  const int ch = blockIdx.x, tid = threadIdx.x;
  float s = 0.f, q = 0.f;
  for (int b = 0; b < kB; ++b) {
    const unsigned short* p = ybf + ((size_t)b * kCHY + ch) * kT;
    for (int i = tid * 8; i < kT; i += 256 * 8) {
      u16x8 v = *(const u16x8*)(p + i);
      #pragma unroll
      for (int k2 = 0; k2 < 8; ++k2) { float f = bf2f(v[k2]); s += f; q += f * f; }
    }
  }
  sred[tid] = s; qred[tid] = q;
  __syncthreads();
  for (int off = 128; off > 0; off >>= 1) {
    if (tid < off) { sred[tid] += sred[tid + off]; qred[tid] += qred[tid + off]; }
    __syncthreads();
  }
  if (tid == 0) {
    const float inv_n = 1.f / (float)(kB * kT);
    float mean = sred[0] * inv_n;
    float var  = qred[0] * inv_n - mean * mean;
    float rs   = rsqrtf(var + kEPS);
    float g    = gamma[ch];
    scale[ch] = g * rs;
    shift[ch] = beta[ch] - mean * g * rs;
  }
}

// ============================================================
// Kernel 3: normalize + edge-pair mean pool + LeakyReLU.
// ============================================================
__global__ __launch_bounds__(256)
void pool_kernel(const unsigned short* __restrict__ ybf,
                 const float* __restrict__ scale, const float* __restrict__ shift,
                 const int* __restrict__ edges, float* __restrict__ out) {
  const int idx = blockIdx.x * 256 + threadIdx.x;   // kB * kP*kCOUT * (kT/8) threads
  const int t8 = (idx & (kT / 8 - 1)) << 3;
  const int pc = (idx >> 10) % (kP * kCOUT);
  const int b  = idx / ((kT / 8) * (kP * kCOUT));
  const int p  = pc >> 5, o = pc & 31;
  const int ch0 = edges[2 * p]     * kCOUT + o;
  const int ch1 = edges[2 * p + 1] * kCOUT + o;
  u16x8 y0 = *(const u16x8*)(ybf + ((size_t)b * kCHY + ch0) * kT + t8);
  u16x8 y1 = *(const u16x8*)(ybf + ((size_t)b * kCHY + ch1) * kT + t8);
  const float s0 = scale[ch0], h0 = shift[ch0];
  const float s1 = scale[ch1], h1 = shift[ch1];
  float* op = out + ((size_t)b * (kP * kCOUT) + pc) * kT + t8;
  f32x4 o0, o1;
  #pragma unroll
  for (int k2 = 0; k2 < 8; ++k2) {
    float z = 0.5f * ((bf2f(y0[k2]) * s0 + h0) + (bf2f(y1[k2]) * s1 + h1));
    float r = (z >= 0.f) ? z : kSLOPE * z;
    if (k2 < 4) o0[k2] = r; else o1[k2 - 4] = r;
  }
  *(f32x4*)(op)     = o0;
  *(f32x4*)(op + 4) = o1;
}

// ============================================================
extern "C" void kernel_launch(void* const* d_in, const int* in_sizes, int n_in,
                              void* d_out, int out_size, void* d_ws, size_t ws_size,
                              hipStream_t stream) {
  const float* x     = (const float*)d_in[0];
  const float* W     = (const float*)d_in[1];
  // d_in[2] (bias): exactly cancelled by BN mean subtraction -> unused.
  const float* gamma = (const float*)d_in[3];
  const float* beta  = (const float*)d_in[4];
  const int*   adj   = (const int*)d_in[5];
  const int*   edges = (const int*)d_in[6];
  float* out = (float*)d_out;

  unsigned short* ybf = (unsigned short*)d_ws;                 // bf16 conv output
  const size_t ybytes = (size_t)kB * kCHY * kT * sizeof(unsigned short); // ~100.7 MB
  float* scale = (float*)((char*)d_ws + ybytes);
  float* shift = scale + kCHY;

  conv_wmma_kernel<<<dim3(kB * kJ * (kT / kTN)), dim3(256), 0, stream>>>(x, W, adj, ybf);
  bn_stats_kernel<<<dim3(kCHY), dim3(256), 0, stream>>>(ybf, gamma, beta, scale, shift);
  pool_kernel<<<dim3((kB * kP * kCOUT * (kT / 8)) / 256), dim3(256), 0, stream>>>(
      ybf, scale, shift, edges, out);
}
